// MultiHeadAttention_34540126994511
// MI455X (gfx1250) — matrix operand order repaired and checked by reading, not compiled, not run on hardware
//
#include <hip/hip_runtime.h>
#include <hip/hip_bf16.h>

// MI455X / gfx1250, wave32. Compute-bound MHA (~137 GFLOP vs ~150MB traffic):
// all matmuls via v_wmma_f32_16x16x32_bf16 (f32 accumulate), softmax in f32.
// GEMM tiles staged by the Tensor Data Mover (TENSORcnt), flash-attention
// K tiles by global_load_async_to_lds (ASYNCcnt); both double-buffered.

typedef __attribute__((ext_vector_type(16))) __bf16 v16bf;
typedef __attribute__((ext_vector_type(8)))  float  v8f;
typedef __attribute__((__vector_size__(4 * sizeof(int)))) int v4i_raw;
typedef __attribute__((__vector_size__(16))) unsigned int v4u_raw;
typedef __attribute__((__vector_size__(32))) int v8i_raw;

#define EMB    1024
#define NHEADS 16
#define HDIM   64
#define SEQ    2048
#define BATCH  4
#define NROWS  (BATCH * SEQ)   // 8192

// ---------------------------------------------------------------------------
// gfx1250 data movers, feature-guarded.
// ---------------------------------------------------------------------------
#if defined(__has_builtin)
#  if __has_builtin(__builtin_amdgcn_global_load_async_to_lds_b128)
#    define HAVE_ASYNC_B128 1
#  endif
#  if __has_builtin(__builtin_amdgcn_tensor_load_to_lds)
#    define HAVE_TDM 1
#  endif
#endif

__device__ __forceinline__ void async_copy16(const __bf16* g, __bf16* l) {
#if defined(HAVE_ASYNC_B128)
    // low 32 bits of a generic LDS address == LDS byte offset (aperture rule)
    __builtin_amdgcn_global_load_async_to_lds_b128(
        (__attribute__((address_space(1))) v4i_raw*)(unsigned long long)g,
        (__attribute__((address_space(3))) v4i_raw*)(unsigned int)(unsigned long long)l,
        0, 0);
#else
    *(uint4*)l = *(const uint4*)g;
#endif
}

__device__ __forceinline__ void async_wait_all() {
#if defined(HAVE_ASYNC_B128)
    asm volatile("s_wait_asynccnt 0x0" ::: "memory");
#endif
}

#if defined(HAVE_TDM)
// 2D TDM tile load: tile_d0 x tile_d1 elements (2-byte), row stride stride0,
// LDS rows padded +4 dwords every 32 dwords (64-half row -> 72-half pitch).
__device__ __forceinline__ void tdm_load_tile_2d(const void* gsrc, void* ldst,
                                                 unsigned tile_d0, unsigned tile_d1,
                                                 unsigned tensor_d0, unsigned tensor_d1,
                                                 unsigned stride0) {
    unsigned long long ga = (unsigned long long)gsrc;
    unsigned lds = (unsigned)(unsigned long long)ldst;
    // D# group 0: count=1 | lds_addr | global_addr[56:0] | type=2
    v4u_raw g0 = { 0x1u, lds, (unsigned)ga,
                   (unsigned)((ga >> 32) & 0x01FFFFFFu) | 0x80000000u };
    // D# group 1: data_size=2B, pad_enable, pad_interval=32dw, pad_amount=4dw,
    //             tensor_dim0/1, tile_dim0/1, tensor_dim0_stride
    v8i_raw g1 = {
        (int)((1u << 16) | (1u << 20) | (4u << 22) | (3u << 25)),
        (int)((tensor_d0 & 0xffffu) << 16),
        (int)((tensor_d0 >> 16) | ((tensor_d1 & 0xffffu) << 16)),
        (int)((tensor_d1 >> 16) | ((tile_d0 & 0xffffu) << 16)),
        (int)(tile_d1 & 0xffffu),
        (int)stride0, 0, 0 };
    v4i_raw z4 = {0, 0, 0, 0};
#if __clang_major__ >= 23
    v8i_raw z8 = {0, 0, 0, 0, 0, 0, 0, 0};
    __builtin_amdgcn_tensor_load_to_lds(g0, g1, z4, z4, z8, 0);
#else
    __builtin_amdgcn_tensor_load_to_lds(g0, g1, z4, z4, 0);
#endif
}
#endif

__device__ __forceinline__ void gemm_stage_wait() {
#if defined(HAVE_TDM)
    __builtin_amdgcn_s_wait_tensorcnt(0);
#elif defined(HAVE_ASYNC_B128)
    asm volatile("s_wait_asynccnt 0x0" ::: "memory");
#endif
}

// ---------------------------------------------------------------------------
// fp32 -> bf16 conversion (bandwidth-trivial: ~50MB total)
// ---------------------------------------------------------------------------
__global__ void cvt_f32_bf16(const float* __restrict__ src,
                             __bf16* __restrict__ dst, int n) {
    int i = blockIdx.x * blockDim.x + threadIdx.x;
    int stride = gridDim.x * blockDim.x;
    for (; i < n; i += stride) dst[i] = (__bf16)src[i];
}

// ---------------------------------------------------------------------------
// bf16 GEMM: C[M x N] = A[M x 1024] * W[N x 1024]^T + bias
// Block tile 128x128, 256 threads = 8 waves (4x2), 32x64 per wave.
// K-step 64, double-buffered LDS, TDM staging, one barrier per K-step.
// LDS row stride 72 halves (36 dwords, gcd(36,64)=4) -> conflict-free b128.
// mode 0: QKV epilogue -> bf16 Q/K/V split per head, [bh][seq][d]
// mode 1: proj epilogue -> fp32 out + bias
// ---------------------------------------------------------------------------
#define LDT_S 72
#define TK    64

__global__ void __launch_bounds__(256)
gemm_bf16(const __bf16* __restrict__ A,
          const __bf16* __restrict__ W,
          const float*  __restrict__ bias,
          int mode,
          __bf16* __restrict__ qb, __bf16* __restrict__ kb,
          __bf16* __restrict__ vb,
          float* __restrict__ outf)
{
    __shared__ __bf16 sA[2][128 * LDT_S];
    __shared__ __bf16 sB[2][128 * LDT_S];

    const int tid  = threadIdx.x;
    const int lane = tid & 31;
    const int wid  = tid >> 5;       // 0..7
    const int wm   = wid >> 1;       // 0..3  -> 32-row slice
    const int wn   = wid & 1;        // 0..1  -> 64-col slice
    const int hl   = lane >> 4;      // half-wave
    const int l16  = lane & 15;

    const int mblk = blockIdx.y * 128;
    const int nblk = blockIdx.x * 128;
    const int K = EMB;
    const int NK = K / TK;           // 16

    // stage tile 0 into buffer 0 (A: 128x64 halves, B: 128x64 halves)
#if defined(HAVE_TDM)
    if (wid == 0) {
        tdm_load_tile_2d(A + (size_t)mblk * K, &sA[0][0], TK, 128, K, 128, K);
        tdm_load_tile_2d(W + (size_t)nblk * K, &sB[0][0], TK, 128, K, 128, K);
    }
#else
    {
        const int srow = tid >> 3, sck = tid & 7;
        #pragma unroll
        for (int it = 0; it < 4; ++it) {
            int row = srow + it * 32;
            async_copy16(A + (size_t)(mblk + row) * K + sck * 8,
                         &sA[0][row * LDT_S + sck * 8]);
            async_copy16(W + (size_t)(nblk + row) * K + sck * 8,
                         &sB[0][row * LDT_S + sck * 8]);
        }
    }
#endif

    v8f acc[2][4] = {};

    for (int kt = 0; kt < NK; ++kt) {
        gemm_stage_wait();
        __syncthreads();                       // buffer kt&1 ready block-wide

        if (kt + 1 < NK) {                     // prefetch next tile
            int k0 = (kt + 1) * TK;
#if defined(HAVE_TDM)
            if (wid == 0) {
                tdm_load_tile_2d(A + (size_t)mblk * K + k0, &sA[(kt + 1) & 1][0],
                                 TK, 128, K, 128, K);
                tdm_load_tile_2d(W + (size_t)nblk * K + k0, &sB[(kt + 1) & 1][0],
                                 TK, 128, K, 128, K);
            }
#else
            {
                const int srow = tid >> 3, sck = tid & 7;
                __bf16* dA = sA[(kt + 1) & 1];
                __bf16* dB = sB[(kt + 1) & 1];
                #pragma unroll
                for (int it = 0; it < 4; ++it) {
                    int row = srow + it * 32;
                    async_copy16(A + (size_t)(mblk + row) * K + k0 + sck * 8,
                                 &dA[row * LDT_S + sck * 8]);
                    async_copy16(W + (size_t)(nblk + row) * K + k0 + sck * 8,
                                 &dB[row * LDT_S + sck * 8]);
                }
            }
#endif
        }

        const __bf16* cA = sA[kt & 1];
        const __bf16* cB = sB[kt & 1];

        #pragma unroll
        for (int kk = 0; kk < 2; ++kk) {       // two 32-wide K slices
            v16bf af[2];
            #pragma unroll
            for (int mi = 0; mi < 2; ++mi) {
                int row = wm * 32 + mi * 16 + l16;
                union { v16bf v; uint4 u[2]; } t;
                t.u[0] = *(const uint4*)(&cA[row * LDT_S + kk * 32 + hl * 8]);
                t.u[1] = *(const uint4*)(&cA[row * LDT_S + kk * 32 + 16 + hl * 8]);
                af[mi] = t.v;
            }
            #pragma unroll
            for (int ni = 0; ni < 4; ++ni) {
                int col = wn * 64 + ni * 16 + l16;
                union { v16bf v; uint4 u[2]; } t;
                t.u[0] = *(const uint4*)(&cB[col * LDT_S + kk * 32 + hl * 8]);
                t.u[1] = *(const uint4*)(&cB[col * LDT_S + kk * 32 + 16 + hl * 8]);
                #pragma unroll
                for (int mi = 0; mi < 2; ++mi)
                    acc[mi][ni] = __builtin_amdgcn_wmma_f32_16x16x32_bf16(
                        false, af[mi], false, t.v, (short)0, acc[mi][ni],
                        false, false);
            }
        }
    }

    // epilogue — C layout: VGPR r, lane -> M = r + 8*(lane>>4), N = lane&15
    #pragma unroll
    for (int mi = 0; mi < 2; ++mi) {
        #pragma unroll
        for (int ni = 0; ni < 4; ++ni) {
            int n = nblk + wn * 64 + ni * 16 + l16;
            float bv = bias[n];
            #pragma unroll
            for (int r = 0; r < 8; ++r) {
                int m = mblk + wm * 32 + mi * 16 + r + 8 * hl;
                float v = acc[mi][ni][r] + bv;
                if (mode == 0) {
                    int which = n >> 10;          // 0=q 1=k 2=v
                    int e = n & 1023;
                    int h = e >> 6, d = e & 63;
                    int b = m >> 11, s = m & 2047;
                    size_t dst = ((size_t)((b * NHEADS + h) * SEQ + s)) * HDIM + d;
                    __bf16* base = (which == 0) ? qb : (which == 1) ? kb : vb;
                    base[dst] = (__bf16)v;
                } else {
                    outf[(size_t)m * EMB + n] = v;
                }
            }
        }
    }
}

// ---------------------------------------------------------------------------
// Flash attention: grid (16 q-tiles, 64 b*h), 256 threads = 8 waves,
// each wave owns 16 query rows. 32 keys/iter, double-buffered K/V staging
// (K async-to-LDS, V manual transpose):
//   S(16x32) = Q(16x64) K^T : 2x2 WMMAs, online softmax, P via LDS re-layout,
//   O(16x64) += P(16x32) V  : 4 WMMAs.
// ---------------------------------------------------------------------------
#define KT     32
#define LDK_S  72   // K-tile row stride (halves): 36 dwords, conflict-free
#define LDV_S  40   // transposed V-tile row stride (20 dwords, conflict-free)
#define LDP_S  40   // per-wave P tile row stride

__global__ void __launch_bounds__(256)
flash_attn(const __bf16* __restrict__ Q,
           const __bf16* __restrict__ Kk,
           const __bf16* __restrict__ V,
           __bf16* __restrict__ attn)          // [8192][1024] bf16
{
    __shared__ __bf16 sK[2][KT * LDK_S];       // [key][d]
    __shared__ __bf16 sV[2][HDIM * LDV_S];     // [d][key]  (transposed)
    __shared__ __bf16 sP[8 * 16 * LDP_S];      // per-wave P tiles

    const int tid  = threadIdx.x;
    const int lane = tid & 31;
    const int wid  = tid >> 5;
    const int hl   = lane >> 4;
    const int l16  = lane & 15;

    const int bh = blockIdx.y;                 // 0..63
    const int b  = bh >> 4, h = bh & 15;
    const int qbase = blockIdx.x * 128 + wid * 16;

    const __bf16* Qb = Q  + (size_t)bh * SEQ * HDIM;
    const __bf16* Kb = Kk + (size_t)bh * SEQ * HDIM;
    const __bf16* Vb = V  + (size_t)bh * SEQ * HDIM;

    const int skey = tid >> 3, sdc = tid & 7;  // staging decomposition

    // Q fragments live in registers for the whole kernel
    v16bf aq[2];
    {
        const __bf16* qr = Qb + (size_t)(qbase + l16) * HDIM;
        #pragma unroll
        for (int kk = 0; kk < 2; ++kk) {
            union { v16bf v; uint4 u[2]; } t;
            t.u[0] = *(const uint4*)(qr + kk * 32 + hl * 8);
            t.u[1] = *(const uint4*)(qr + kk * 32 + 16 + hl * 8);
            aq[kk] = t.v;
        }
    }

    // stage tile 0: K async, V manual transpose
    {
        async_copy16(Kb + (size_t)skey * HDIM + sdc * 8,
                     &sK[0][skey * LDK_S + sdc * 8]);
        union { uint4 u; __bf16 e[8]; } vv;
        vv.u = *(const uint4*)(Vb + (size_t)skey * HDIM + sdc * 8);
        #pragma unroll
        for (int i = 0; i < 8; ++i)
            sV[0][(sdc * 8 + i) * LDV_S + skey] = vv.e[i];
    }

    v8f o[4] = {};
    float rm[8], rl[8];
    #pragma unroll
    for (int r = 0; r < 8; ++r) { rm[r] = -1e30f; rl[r] = 0.0f; }

    const int NJ = SEQ / KT;                   // 64
    for (int jt = 0; jt < NJ; ++jt) {
        async_wait_all();
        __syncthreads();                       // buffer jt&1 ready

        if (jt + 1 < NJ) {                     // prefetch next K/V tile
            int j = (jt + 1) * KT;
            __bf16* dK = sK[(jt + 1) & 1];
            __bf16* dV = sV[(jt + 1) & 1];
            async_copy16(Kb + (size_t)(j + skey) * HDIM + sdc * 8,
                         &dK[skey * LDK_S + sdc * 8]);
            union { uint4 u; __bf16 e[8]; } vv;
            vv.u = *(const uint4*)(Vb + (size_t)(j + skey) * HDIM + sdc * 8);
            #pragma unroll
            for (int i = 0; i < 8; ++i)
                dV[(sdc * 8 + i) * LDV_S + skey] = vv.e[i];
        }

        const __bf16* cK = sK[jt & 1];
        const __bf16* cV = sV[jt & 1];

        // S = Q K^T over two 16-key column tiles, d split 0..31 / 32..63
        v8f s[2];
        #pragma unroll
        for (int nt = 0; nt < 2; ++nt) {
            v8f sacc = {};
            int keycol = nt * 16 + l16;
            #pragma unroll
            for (int kk = 0; kk < 2; ++kk) {
                union { v16bf v; uint4 u[2]; } t;
                t.u[0] = *(const uint4*)(&cK[keycol * LDK_S + kk * 32 + hl * 8]);
                t.u[1] = *(const uint4*)(&cK[keycol * LDK_S + kk * 32 + 16 + hl * 8]);
                sacc = __builtin_amdgcn_wmma_f32_16x16x32_bf16(
                    false, aq[kk], false, t.v, (short)0, sacc, false, false);
            }
            s[nt] = sacc;
        }

        const float scale = 0.125f;  // 1/sqrt(64)
        __bf16* pbase = &sP[wid * 16 * LDP_S];

        // online softmax, row-wise (row = r + 8*hl, cols across 16 lanes)
        #pragma unroll
        for (int r = 0; r < 8; ++r) {
            float s0 = s[0][r] * scale, s1 = s[1][r] * scale;
            float t = fmaxf(s0, s1);
            #pragma unroll
            for (int off = 1; off < 16; off <<= 1)
                t = fmaxf(t, __shfl_xor(t, off));
            float mnew  = fmaxf(rm[r], t);
            float alpha = __expf(rm[r] - mnew);
            float p0 = __expf(s0 - mnew);
            float p1 = __expf(s1 - mnew);
            float ps = p0 + p1;
            #pragma unroll
            for (int off = 1; off < 16; off <<= 1)
                ps += __shfl_xor(ps, off);
            rl[r] = rl[r] * alpha + ps;
            rm[r] = mnew;
            #pragma unroll
            for (int ni = 0; ni < 4; ++ni) o[ni][r] *= alpha;
            // C-layout -> LDS [qrow][key] (bf16) for A-layout reload
            pbase[(r + 8 * hl) * LDP_S + l16]      = (__bf16)p0;
            pbase[(r + 8 * hl) * LDP_S + 16 + l16] = (__bf16)p1;
        }

        // reload P as A fragment (wave-private; DS ops are in-order per wave)
        v16bf pf;
        {
            union { v16bf v; uint4 u[2]; } t;
            t.u[0] = *(const uint4*)(pbase + l16 * LDP_S + hl * 8);
            t.u[1] = *(const uint4*)(pbase + l16 * LDP_S + 16 + hl * 8);
            pf = t.v;
        }

        // O += P @ V   (B = [key][d] from transposed V tile)
        #pragma unroll
        for (int ni = 0; ni < 4; ++ni) {
            int dcol = ni * 16 + l16;
            union { v16bf v; uint4 u[2]; } t;
            t.u[0] = *(const uint4*)(&cV[dcol * LDV_S + hl * 8]);
            t.u[1] = *(const uint4*)(&cV[dcol * LDV_S + 16 + hl * 8]);
            o[ni] = __builtin_amdgcn_wmma_f32_16x16x32_bf16(
                false, pf, false, t.v, (short)0, o[ni], false, false);
        }
    }

    // normalize + write bf16 attention output [m][h*64+d]
    #pragma unroll
    for (int r = 0; r < 8; ++r) {
        float inv = 1.0f / rl[r];
        int qrow = qbase + r + 8 * hl;
        size_t mrow = (size_t)(b * SEQ + qrow);
        #pragma unroll
        for (int ni = 0; ni < 4; ++ni) {
            int d = ni * 16 + l16;
            attn[mrow * EMB + h * HDIM + d] = (__bf16)(o[ni][r] * inv);
        }
    }
}

// ---------------------------------------------------------------------------
// launch
// ---------------------------------------------------------------------------
extern "C" void kernel_launch(void* const* d_in, const int* in_sizes, int n_in,
                              void* d_out, int out_size, void* d_ws, size_t ws_size,
                              hipStream_t stream)
{
    const float* x      = (const float*)d_in[0];
    const float* w_qkv  = (const float*)d_in[1];
    const float* b_qkv  = (const float*)d_in[2];
    const float* w_proj = (const float*)d_in[3];
    const float* b_proj = (const float*)d_in[4];
    float* out = (float*)d_out;

    char* p = (char*)d_ws;
    __bf16* xb     = (__bf16*)p; p += (size_t)NROWS * EMB * 2;       // 16 MB
    __bf16* wqkvb  = (__bf16*)p; p += (size_t)3 * EMB * EMB * 2;     //  6 MB
    __bf16* wprojb = (__bf16*)p; p += (size_t)EMB * EMB * 2;         //  2 MB
    __bf16* qb     = (__bf16*)p; p += (size_t)NROWS * EMB * 2;       // 16 MB
    __bf16* kb     = (__bf16*)p; p += (size_t)NROWS * EMB * 2;       // 16 MB
    __bf16* vb     = (__bf16*)p; p += (size_t)NROWS * EMB * 2;       // 16 MB
    __bf16* attnb  = (__bf16*)p;                                     // 16 MB

    cvt_f32_bf16<<<2048, 256, 0, stream>>>(x, xb, NROWS * EMB);
    cvt_f32_bf16<<<1024, 256, 0, stream>>>(w_qkv, wqkvb, 3 * EMB * EMB);
    cvt_f32_bf16<<<512, 256, 0, stream>>>(w_proj, wprojb, EMB * EMB);

    // QKV: M=8192, N=3072 -> grid (24, 64)
    gemm_bf16<<<dim3(24, 64), 256, 0, stream>>>(
        xb, wqkvb, b_qkv, 0, qb, kb, vb, nullptr);

    // attention: 16 query tiles x 64 (b*h)
    flash_attn<<<dim3(16, 64), 256, 0, stream>>>(qb, kb, vb, attnb);

    // proj: M=8192, N=1024 -> grid (8, 64)
    gemm_bf16<<<dim3(8, 64), 256, 0, stream>>>(
        attnb, wprojb, b_proj, 1, nullptr, nullptr, nullptr, out);
}
